// Model_66614942761126
// MI455X (gfx1250) — compile-verified, matrix operand order checked by reading, not run on hardware
//
#include <hip/hip_runtime.h>
#include <hip/hip_bf16.h>
#include <math.h>

// ---------------------------------------------------------------------------
// Model constants (from the reference)
// ---------------------------------------------------------------------------
#define ROWSC 16
#define COLSC 16
#define CHC   3
#define HIDC  512
#define HEADSC 8
#define DHC   64          // HID / HEADS
#define BLENC 128
#define SEQC  768         // ROWS*COLS*CH
#define NBC   6           // SEQ / BLEN
#define BATCHC 16
#define TOKC  (BATCHC * SEQC)   // 12288
#define NFC   256
#define LYRC  6
#define EPSC  1e-6f

typedef _Float16 h16_t;
typedef __attribute__((ext_vector_type(16))) _Float16 v16h;
typedef __attribute__((ext_vector_type(8)))  float    v8f;
typedef __attribute__((ext_vector_type(4)))  unsigned v4u;
typedef __attribute__((ext_vector_type(8)))  int      v8i;
typedef __attribute__((ext_vector_type(4)))  int      v4i;

#if __has_builtin(__builtin_amdgcn_tensor_load_to_lds)
#define HAVE_TDM 1
#endif

__device__ __forceinline__ v8f zero8() {
  v8f z = {0.f, 0.f, 0.f, 0.f, 0.f, 0.f, 0.f, 0.f};
  return z;
}

__device__ __forceinline__ v8f wmma_f16(v16h a, v16h b, v8f c) {
  // v_wmma_f32_16x16x32_f16  D = A*B + C   (wave32, f32 accum)
  return __builtin_amdgcn_wmma_f32_16x16x32_f16(
      /*neg_a=*/false, a, /*neg_b=*/false, b,
      /*c_mod=*/(short)0, c, /*reuse_a=*/false, /*reuse_b=*/false);
}

// ---------------------------------------------------------------------------
// CDNA5 async global->LDS copy (per-lane, tracked by ASYNCcnt).
// Portable inline asm per CDNA5 ISA (GV mode).
// ---------------------------------------------------------------------------
__device__ __forceinline__ void async_copy_b128(const void* gptr, void* lptr) {
  unsigned lds_off = (unsigned)(unsigned long long)(uintptr_t)lptr; // low 32 = LDS addr
  asm volatile("global_load_async_to_lds_b128 %0, %1, off"
               :
               : "v"(lds_off), "v"(gptr)
               : "memory");
}
__device__ __forceinline__ void wait_async0() {
  asm volatile("s_wait_asynccnt 0x0" ::: "memory");
}
// s_wait_tensorcnt takes an immediate: one wrapper per literal value.
__device__ __forceinline__ void wait_tensor0() {
#if __has_builtin(__builtin_amdgcn_s_wait_tensorcnt)
  __builtin_amdgcn_s_wait_tensorcnt(0);
#else
  asm volatile("s_wait_tensorcnt 0x0" ::: "memory");
#endif
}
__device__ __forceinline__ void wait_tensor2() {
#if __has_builtin(__builtin_amdgcn_s_wait_tensorcnt)
  __builtin_amdgcn_s_wait_tensorcnt(2);
#else
  asm volatile("s_wait_tensorcnt 0x2" ::: "memory");
#endif
}

#ifdef HAVE_TDM
// Issue a TDM load of a tile_rows x 32-half f16 tile (row stride = ld halves)
// into LDS with 16-DWORD data / 4-DWORD pad interleave (matches LDS stride of
// 40 halves).  Wave-level instruction: call from one wave only.
__device__ __forceinline__ void tdm_load_tile(const h16_t* gbase, int ld,
                                              int tot_rows, int tile_rows,
                                              unsigned lds_off) {
  unsigned long long ga = (unsigned long long)(uintptr_t)gbase;
  v4u g0;
  g0[0] = 1u;                                   // count=1 valid descriptor
  g0[1] = lds_off;                              // lds_addr (bytes)
  g0[2] = (unsigned)(ga & 0xFFFFFFFFull);       // global_addr[31:0]
  g0[3] = (unsigned)((ga >> 32) & 0x1FFFFFFull) | (2u << 30);  // addr[56:32]|type=2
  v8i g1;
  // data_size=1(2B) | pad_enable | pad_interval=3(16DW) | pad_amount=3(4DW)
  g1[0] = (int)((1u << 16) | (1u << 20) | (3u << 22) | (3u << 25));
  g1[1] = (int)(((unsigned)ld & 0xFFFFu) << 16);                 // tensor_dim0 lo
  g1[2] = (int)((((unsigned)ld >> 16) & 0xFFFFu) |               // tensor_dim0 hi
                (((unsigned)tot_rows & 0xFFFFu) << 16));         // tensor_dim1 lo
  g1[3] = (int)((((unsigned)tot_rows >> 16) & 0xFFFFu) | (32u << 16)); // |tile_dim0
  g1[4] = (int)(unsigned)tile_rows;                              // tile_dim1
  g1[5] = (int)(unsigned)ld;                                     // dim0_stride lo
  g1[6] = 0;                                                     // stride hi
  g1[7] = 0;
  v4i zz = {0, 0, 0, 0};
#if __clang_major__ >= 23
  v8i z8 = {0, 0, 0, 0, 0, 0, 0, 0};
  __builtin_amdgcn_tensor_load_to_lds(g0, g1, zz, zz, z8, 0);
#else
  __builtin_amdgcn_tensor_load_to_lds(g0, g1, zz, zz, 0);
#endif
}
#endif

#define LDT 40   // padded LDS row stride (halves) for 32-half tile rows

// Fallback: per-lane async staging of a rows x 32-half tile into padded LDS.
__device__ __forceinline__ void stage_async_tile(const h16_t* g, int ld,
                                                 h16_t* lds, int rows) {
  for (int u = threadIdx.x; u < rows * 4; u += (int)blockDim.x) {
    int r = u >> 2, c8 = (u & 3) * 8;
    async_copy_b128(g + (size_t)r * ld + c8, lds + r * LDT + c8);
  }
}

// A-matrix 16x32 f16 fragment (also works for B^T stored row-major, since the
// B layout mirrors A with N in place of M).  Rows m0..m0+15 (lane-mapped),
// cols k0..k0+31 (packed per ISA 7.12.2).
__device__ __forceinline__ v16h frag_rowmajor(const h16_t* __restrict__ base,
                                              int ld, int m0, int k0) {
  const int lane = threadIdx.x & 31;
  const int hs = lane >> 4;
  const h16_t* p = base + (size_t)(m0 + (lane & 15)) * ld + k0 + hs * 8;
  v16h f;
#pragma unroll
  for (int j = 0; j < 8; ++j) { f[j] = p[j]; f[8 + j] = p[16 + j]; }
  return f;
}

// B-matrix fragment from a row-major K x N *f32* matrix with in-register
// f32 -> f16 conversion (proj_W streams HBM exactly once, straight to WMMA).
__device__ __forceinline__ v16h frag_b_f32(const float* __restrict__ base,
                                           int ld, int k0, int n0) {
  const int lane = threadIdx.x & 31;
  const int hs = lane >> 4;
  const int c = n0 + (lane & 15);
  const int kb = k0 + hs * 8;
  v16h f;
#pragma unroll
  for (int j = 0; j < 8; ++j) {
    f[j]     = (h16_t)base[(size_t)(kb + j) * ld + c];
    f[8 + j] = (h16_t)base[(size_t)(kb + 16 + j) * ld + c];
  }
  return f;
}

__device__ __forceinline__ float gelu_exact(float x) {
  return 0.5f * x * (1.f + erff(x * 0.70710678118654752f));
}

// ---------------------------------------------------------------------------
// f32 (K x N) -> f16 transposed (N x K) weight conversion.  Paid once per
// layer; lets every GEMM B tile be a plain strided 2-D tile for the TDM.
// ---------------------------------------------------------------------------
__global__ __launch_bounds__(256) void k_f32_to_f16_T(const float* __restrict__ src,
                                                      h16_t* __restrict__ dst,
                                                      int K, int N) {
  int kk = blockIdx.x * 32 + (threadIdx.x & 31);
  int nn = blockIdx.y * 8 + (threadIdx.x >> 5);
  dst[(size_t)nn * K + kk] = (h16_t)src[(size_t)kk * N + nn];
}

// ---------------------------------------------------------------------------
// Embedding gather + sequence shift + timing signal
// ---------------------------------------------------------------------------
__device__ __forceinline__ float timing_val(int r, int cc, int c) {
  const float kexp = -9.210340371976184f / 127.f;   // -ln(10000)/(nts-1)
  if (c < 256) {
    int i = c & 127;
    float ang = (float)r * __expf(kexp * (float)i);
    return (c < 128) ? __sinf(ang) : __cosf(ang);
  } else {
    int i = (c - 256) & 127;
    float ang = (float)cc * __expf(kexp * (float)i);
    return (c < 384) ? __sinf(ang) : __cosf(ang);
  }
}

__global__ __launch_bounds__(256) void k_embed(const float* __restrict__ X,
                                               const float* __restrict__ tbl,
                                               float* __restrict__ h,
                                               h16_t* __restrict__ hh) {
  const int token = blockIdx.x;             // 0..TOK-1
  const int b = token / SEQC, s = token % SEQC;
  const int r = s / (COLSC * CHC), cc = s % (COLSC * CHC);
  const int sp = s - 1;                     // shifted source position
  const bool valid = (sp >= 0);
  int idx = 0;
  if (valid) {
    int ccp = sp % (COLSC * CHC);
    int rp  = sp / (COLSC * CHC);
    int colp = ccp / CHC, chp = ccp % CHC;
    float xv = X[((b * CHC + chp) * ROWSC + rp) * COLSC + colp];
    idx = (int)(xv * 255.f) + chp * 256;    // (NUM_POINTS-1)=255, chan_add
  }
#pragma unroll
  for (int half = 0; half < 2; ++half) {
    int c = (int)threadIdx.x + half * 256;
    float e = valid ? tbl[(size_t)idx * HIDC + c] * 22.62741699796952f : 0.f;
    float v = e + timing_val(r, cc, c);
    h [(size_t)token * HIDC + c] = v;
    hh[(size_t)token * HIDC + c] = (h16_t)v;
  }
}

// ---------------------------------------------------------------------------
// Tiled WMMA GEMM: C(MxN) = A(MxK,f16) * BT(NxK,f16)^T with fused epilogues.
//   mode 0: f32 store (+bias)                (attn-out @ Wo, MLP2)
//   mode 1: f16 store of gelu(acc+bias)      (MLP1)
//   mode 2: f16 head-split (B,H,S,dh)*scale  (Q, K)
//   mode 3: f16 head-split (B,H,dh,S)*scale  (V transposed for P@V B-frags)
// Block tile 128x64, 256 threads (8 waves), K step 32.
// Both tiles are DMA'd by the Tensor Data Mover (pad fields generate the
// padded LDS layout) with double buffering: next K-step's loads are issued
// before waiting on the current pair (per-wave in-order TDM completion).
// ---------------------------------------------------------------------------
#define BM 128
#define BN 64
#define BK 32

__global__ __launch_bounds__(256, 2) void k_gemm(const h16_t* __restrict__ A,
                                                 const h16_t* __restrict__ BT,
                                                 const float* __restrict__ bias,
                                                 float* __restrict__ Cf,
                                                 h16_t* __restrict__ Chh,
                                                 int M, int N, int K,
                                                 int mode, float scale) {
  __shared__ h16_t At[2][BM][LDT];   // 2 x 10 KB
  __shared__ h16_t Bt[2][BN][LDT];   // 2 x  5 KB

  const int tid = threadIdx.x;
  const int wv = tid >> 5;
  const int m0 = blockIdx.x * BM;
  const int n0 = blockIdx.y * BN;
  const h16_t* Ab = A  + (size_t)m0 * K;
  const h16_t* Bb = BT + (size_t)n0 * K;

  v8f acc[4] = {zero8(), zero8(), zero8(), zero8()};

  // prologue: DMA first pair of tiles into buffer 0
#ifdef HAVE_TDM
  if (wv == 0) {
    tdm_load_tile(Ab, K, M, BM, (unsigned)(unsigned long long)(uintptr_t)&At[0][0][0]);
    tdm_load_tile(Bb, K, N, BN, (unsigned)(unsigned long long)(uintptr_t)&Bt[0][0][0]);
  }
#else
  stage_async_tile(Ab, K, &At[0][0][0], BM);
  stage_async_tile(Bb, K, &Bt[0][0][0], BN);
#endif

  const int nsteps = K / BK;
  for (int s = 0; s < nsteps; ++s) {
    const int cur = s & 1;
    const bool more = (s + 1) < nsteps;
    const int knext = (s + 1) * BK;
#ifdef HAVE_TDM
    if (wv == 0) {
      if (more) {   // issue next pair before waiting on current pair
        tdm_load_tile(Ab + knext, K, M, BM,
                      (unsigned)(unsigned long long)(uintptr_t)&At[cur ^ 1][0][0]);
        tdm_load_tile(Bb + knext, K, N, BN,
                      (unsigned)(unsigned long long)(uintptr_t)&Bt[cur ^ 1][0][0]);
        wait_tensor2();   // current pair has landed (per-wave in-order TDM)
      } else {
        wait_tensor0();
      }
    }
#else
    if (more) {
      stage_async_tile(Ab + knext, K, &At[cur ^ 1][0][0], BM);
      stage_async_tile(Bb + knext, K, &Bt[cur ^ 1][0][0], BN);
    }
    wait_async0();
#endif
    __syncthreads();

    v16h af = frag_rowmajor(&At[cur][0][0], LDT, wv * 16, 0);
#pragma unroll
    for (int nt = 0; nt < 4; ++nt) {
      v16h bf = frag_rowmajor(&Bt[cur][0][0], LDT, nt * 16, 0);
      acc[nt] = wmma_f16(af, bf, acc[nt]);
    }
    __syncthreads();
  }

  // --- epilogue (C layout: VGPR r -> row m0+w*16+r+8*hs, lane&15 -> col) ---
  const int lane = tid & 31;
  const int hs = lane >> 4;
  const int cn = lane & 15;
#pragma unroll
  for (int nt = 0; nt < 4; ++nt) {
#pragma unroll
    for (int r = 0; r < 8; ++r) {
      int row = m0 + wv * 16 + r + hs * 8;
      int col = n0 + nt * 16 + cn;
      float v = acc[nt][r];
      if (bias) v += bias[col];
      if (mode == 0) {
        Cf[(size_t)row * N + col] = v;
      } else if (mode == 1) {
        Chh[(size_t)row * N + col] = (h16_t)gelu_exact(v);
      } else {
        v *= scale;
        int bb = row / SEQC, s2 = row % SEQC;
        int hhd = col / DHC, d = col % DHC;
        if (mode == 2)
          Chh[((size_t)(bb * HEADSC + hhd) * SEQC + s2) * DHC + d] = (h16_t)v;
        else
          Chh[((size_t)(bb * HEADSC + hhd) * DHC + d) * SEQC + s2] = (h16_t)v;
      }
    }
  }
}

// ---------------------------------------------------------------------------
// Block-local attention.  grid = (NB, HEADS, BATCH), 128 threads (4 waves).
// K tile (up to 256x64 f16, shared by all 8 Q-strips) is staged into LDS via
// async global->LDS DMA; logits (16 x CL) live in registers, softmax via
// 16-lane shfl_xor reductions, probs bounced through wave-private LDS to
// convert C-layout -> A-layout, then WMMA P @ V.
// ---------------------------------------------------------------------------
__global__ __launch_bounds__(128, 1) void k_attn(const h16_t* __restrict__ q,
                                                 const h16_t* __restrict__ k,
                                                 const h16_t* __restrict__ vT,
                                                 h16_t* __restrict__ ao) {
  __shared__ h16_t Kt[2 * BLENC][DHC];       // 32 KB
  __shared__ h16_t pbuf[4][16][2 * BLENC];   // 32 KB (wave-private strips)

  const int n = blockIdx.x, hh = blockIdx.y, b = blockIdx.z;
  const int wv = threadIdx.x >> 5;
  const int lane = threadIdx.x & 31;
  const int hs = lane >> 4;
  const int cn = lane & 15;

  const int CL = (n == 0) ? BLENC : 2 * BLENC;
  const int kvb = (n == 0) ? 0 : (n - 1) * BLENC;
  const int ntile = CL / 16;
  const int nkt = CL / 32;

  const h16_t* qp = q  + (size_t)(b * HEADSC + hh) * SEQC * DHC; // [S][dh]
  const h16_t* kp = k  + (size_t)(b * HEADSC + hh) * SEQC * DHC; // [S][dh]
  const h16_t* vp = vT + (size_t)(b * HEADSC + hh) * DHC * SEQC; // [dh][S]

  // ---- async-DMA the K tile into LDS (CL x 64 halves, contiguous rows) ----
  {
    const int nchunk = (CL * DHC) / 8;       // 16-byte chunks
    for (int u = threadIdx.x; u < nchunk; u += 128) {
      int row = u >> 3, c8 = (u & 7) * 8;
      async_copy_b128(kp + (size_t)(kvb + row) * DHC + c8, &Kt[row][c8]);
    }
    wait_async0();
  }
  __syncthreads();

  for (int sp = wv; sp < 8; sp += 4) {
    const int qr = n * BLENC + sp * 16;       // first Q row of this strip
    v16h aq0 = frag_rowmajor(qp, DHC, qr, 0);
    v16h aq1 = frag_rowmajor(qp, DHC, qr, 32);

    // ---- logits = Q @ K^T  (K tile in LDS, [key][dh] == B^T row-major) ---
    v8f lg[16];
#pragma unroll
    for (int t = 0; t < 16; ++t) {
      if (t < ntile) {
        v16h bk0 = frag_rowmajor(&Kt[0][0], DHC, t * 16, 0);
        v16h bk1 = frag_rowmajor(&Kt[0][0], DHC, t * 16, 32);
        v8f a = zero8();
        a = wmma_f16(aq0, bk0, a);
        a = wmma_f16(aq1, bk1, a);
        lg[t] = a;
      }
    }

    // ---- causal mask + softmax (rows spread across 16-lane halves) -------
#pragma unroll
    for (int r = 0; r < 8; ++r) {
      const int qpos = qr + r + hs * 8;
      float mx = -3.0e38f;
#pragma unroll
      for (int t = 0; t < 16; ++t) {
        if (t < ntile) {
          int kpos = kvb + t * 16 + cn;
          float val = lg[t][r] + ((kpos > qpos) ? -1.0e9f : 0.f);
          lg[t][r] = val;
          mx = fmaxf(mx, val);
        }
      }
#pragma unroll
      for (int m = 1; m < 16; m <<= 1) mx = fmaxf(mx, __shfl_xor(mx, m, 32));
      float sum = 0.f;
#pragma unroll
      for (int t = 0; t < 16; ++t) {
        if (t < ntile) {
          float e = __expf(lg[t][r] - mx);
          lg[t][r] = e;
          sum += e;
        }
      }
#pragma unroll
      for (int m = 1; m < 16; m <<= 1) sum += __shfl_xor(sum, m, 32);
      float inv = 1.f / sum;
#pragma unroll
      for (int t = 0; t < 16; ++t)
        if (t < ntile)
          pbuf[wv][r + hs * 8][t * 16 + cn] = (h16_t)(lg[t][r] * inv);
    }

    // ---- out = P @ V  (V^T in [dh][S] layout -> contiguous B^T frags) ----
    v8f oa[4] = {zero8(), zero8(), zero8(), zero8()};
#pragma unroll
    for (int kt = 0; kt < 8; ++kt) {
      if (kt < nkt) {
        v16h pa = frag_rowmajor(&pbuf[wv][0][0], 2 * BLENC, 0, kt * 32);
#pragma unroll
        for (int nt = 0; nt < 4; ++nt) {
          v16h vb = frag_rowmajor(vp, SEQC, nt * 16, kvb + kt * 32);
          oa[nt] = wmma_f16(pa, vb, oa[nt]);
        }
      }
    }
    // write attn output in (token, HID) layout for the Wo GEMM
#pragma unroll
    for (int nt = 0; nt < 4; ++nt)
#pragma unroll
      for (int r = 0; r < 8; ++r) {
        int row = b * SEQC + qr + r + hs * 8;
        int col = hh * DHC + nt * 16 + cn;
        ao[(size_t)row * HIDC + col] = (h16_t)oa[nt][r];
      }
  }
}

// ---------------------------------------------------------------------------
// h = LayerNorm(y + h), also refresh the f16 copy.  One token per block.
// ---------------------------------------------------------------------------
__global__ __launch_bounds__(256) void k_add_ln(const float* __restrict__ y,
                                                float* __restrict__ h,
                                                h16_t* __restrict__ hh,
                                                const float* __restrict__ g,
                                                const float* __restrict__ bta) {
  __shared__ float red[2][8];
  const int row = blockIdx.x;
  const int t = threadIdx.x;
  float v0 = y[(size_t)row * HIDC + t]       + h[(size_t)row * HIDC + t];
  float v1 = y[(size_t)row * HIDC + t + 256] + h[(size_t)row * HIDC + t + 256];
  float s = v0 + v1, sq = v0 * v0 + v1 * v1;
#pragma unroll
  for (int m = 1; m < 32; m <<= 1) {
    s += __shfl_xor(s, m, 32);
    sq += __shfl_xor(sq, m, 32);
  }
  if ((t & 31) == 0) { red[0][t >> 5] = s; red[1][t >> 5] = sq; }
  __syncthreads();
  float ts = 0.f, tq = 0.f;
#pragma unroll
  for (int i = 0; i < 8; ++i) { ts += red[0][i]; tq += red[1][i]; }
  float mean = ts * (1.f / HIDC);
  float var = tq * (1.f / HIDC) - mean * mean;
  float rstd = rsqrtf(var + EPSC);
  float o0 = (v0 - mean) * rstd * g[t] + bta[t];
  float o1 = (v1 - mean) * rstd * g[t + 256] + bta[t + 256];
  h [(size_t)row * HIDC + t]        = o0;
  h [(size_t)row * HIDC + t + 256]  = o1;
  hh[(size_t)row * HIDC + t]        = (h16_t)o0;
  hh[(size_t)row * HIDC + t + 256]  = (h16_t)o1;
}

// ---------------------------------------------------------------------------
// Final projection: (16 x 393216) @ (393216 x 256), split-K WMMA.
// grid = (4 N-blocks, 768 K-chunks of 512), 128 threads (4 waves, one N-tile
// of 16 each).  A = h in f16 (rows = batch, contiguous K), B = proj_W f32
// converted in-register.  Partials go to a staging buffer (deterministic).
// ---------------------------------------------------------------------------
#define KCHUNK 512
__global__ __launch_bounds__(128, 2) void k_proj(const h16_t* __restrict__ A,
                                                 const float* __restrict__ W,
                                                 float* __restrict__ stage) {
  const int KTOT = SEQC * HIDC;           // 393216
  const int wv = threadIdx.x >> 5;
  const int n0 = blockIdx.x * 64 + wv * 16;
  const int kbase = blockIdx.y * KCHUNK;
  v8f acc = zero8();
  for (int ks = 0; ks < KCHUNK; ks += 32) {
    v16h af = frag_rowmajor(A, KTOT, 0, kbase + ks);  // rows = the 16 batches
    v16h bf = frag_b_f32(W, NFC, kbase + ks, n0);     // f32 -> f16 in-register
    acc = wmma_f16(af, bf, acc);
  }
  const int lane = threadIdx.x & 31;
  const int hs = lane >> 4;
  const int cn = lane & 15;
#pragma unroll
  for (int r = 0; r < 8; ++r) {
    int row = r + hs * 8;                 // batch index 0..15
    int col = n0 + cn;
    stage[((size_t)blockIdx.y * 16 + row) * NFC + col] = acc[r];
  }
}

__global__ __launch_bounds__(256) void k_reduce_stage(const float* __restrict__ stage,
                                                      float* __restrict__ pre) {
  int e = blockIdx.x * blockDim.x + threadIdx.x;     // 0..4095
  float s = 0.f;
  for (int c = 0; c < 768; ++c) s += stage[(size_t)c * 4096 + e];
  pre[e] = s;
}

// Final GELU + LayerNorm over NF=256.  One batch row per block.
__global__ __launch_bounds__(256) void k_out_ln(const float* __restrict__ pre,
                                                const float* __restrict__ g,
                                                const float* __restrict__ bta,
                                                float* __restrict__ out) {
  __shared__ float red[2][8];
  const int row = blockIdx.x;
  const int t = threadIdx.x;
  float x = gelu_exact(pre[(size_t)row * NFC + t]);
  float s = x, sq = x * x;
#pragma unroll
  for (int m = 1; m < 32; m <<= 1) {
    s += __shfl_xor(s, m, 32);
    sq += __shfl_xor(sq, m, 32);
  }
  if ((t & 31) == 0) { red[0][t >> 5] = s; red[1][t >> 5] = sq; }
  __syncthreads();
  float ts = 0.f, tq = 0.f;
#pragma unroll
  for (int i = 0; i < 8; ++i) { ts += red[0][i]; tq += red[1][i]; }
  float mean = ts * (1.f / NFC);
  float var = tq * (1.f / NFC) - mean * mean;
  float rstd = rsqrtf(var + EPSC);
  out[(size_t)row * NFC + t] = (x - mean) * rstd * g[t] + bta[t];
}

// ---------------------------------------------------------------------------
// Host orchestration
// ---------------------------------------------------------------------------
extern "C" void kernel_launch(void* const* d_in, const int* in_sizes, int n_in,
                              void* d_out, int out_size, void* d_ws, size_t ws_size,
                              hipStream_t stream) {
  const float* X    = (const float*)d_in[0];
  const float* tbl  = (const float*)d_in[1];
  const float* Wq   = (const float*)d_in[2];
  const float* Wk   = (const float*)d_in[3];
  const float* Wv   = (const float*)d_in[4];
  const float* Wo   = (const float*)d_in[5];
  const float* ln1g = (const float*)d_in[6];
  const float* ln1b = (const float*)d_in[7];
  const float* W1   = (const float*)d_in[8];
  const float* b1   = (const float*)d_in[9];
  const float* W2   = (const float*)d_in[10];
  const float* b2   = (const float*)d_in[11];
  const float* ln2g = (const float*)d_in[12];
  const float* ln2b = (const float*)d_in[13];
  const float* projW= (const float*)d_in[14];
  const float* olng = (const float*)d_in[15];
  const float* olnb = (const float*)d_in[16];
  float* out = (float*)d_out;

  char* ws = (char*)d_ws;
  size_t off = 0;
  auto carve = [&](size_t bytes) -> void* {
    void* p = ws + off;
    off = (off + bytes + 255) & ~(size_t)255;
    return p;
  };

  float* hf32   = (float*)carve((size_t)TOKC * HIDC * 4);
  float* yf32   = (float*)carve((size_t)TOKC * HIDC * 4);
  h16_t* h16    = (h16_t*)carve((size_t)TOKC * HIDC * 2);
  h16_t* q16    = (h16_t*)carve((size_t)TOKC * HIDC * 2);
  h16_t* k16    = (h16_t*)carve((size_t)TOKC * HIDC * 2);
  h16_t* vT16   = (h16_t*)carve((size_t)TOKC * HIDC * 2);
  h16_t* ao16   = (h16_t*)carve((size_t)TOKC * HIDC * 2);
  h16_t* mid16  = (h16_t*)carve((size_t)TOKC * 4 * HIDC * 2);
  h16_t* wq16   = (h16_t*)carve((size_t)HIDC * HIDC * 2);   // transposed (NxK)
  h16_t* wk16   = (h16_t*)carve((size_t)HIDC * HIDC * 2);
  h16_t* wv16   = (h16_t*)carve((size_t)HIDC * HIDC * 2);
  h16_t* wo16   = (h16_t*)carve((size_t)HIDC * HIDC * 2);
  h16_t* w116   = (h16_t*)carve((size_t)HIDC * 4 * HIDC * 2);
  h16_t* w216   = (h16_t*)carve((size_t)HIDC * 4 * HIDC * 2);
  float* stage  = (float*)carve((size_t)768 * 16 * NFC * 4);
  float* pre    = (float*)carve((size_t)16 * NFC * 4);
  (void)ws_size; (void)in_sizes; (void)n_in; (void)out_size;

  auto convT = [&](const float* s, h16_t* d, int K, int N) {
    k_f32_to_f16_T<<<dim3(K / 32, N / 8), dim3(256), 0, stream>>>(s, d, K, N);
  };
  auto gemm = [&](const h16_t* A, const h16_t* BT, const float* bias,
                  float* Cf, h16_t* Ch, int M, int N, int K, int mode, float sc) {
    k_gemm<<<dim3(M / BM, N / BN), dim3(256), 0, stream>>>(A, BT, bias, Cf, Ch,
                                                           M, N, K, mode, sc);
  };

  // ---- embedding + shift + timing signal --------------------------------
  k_embed<<<dim3(TOKC), dim3(256), 0, stream>>>(X, tbl, hf32, h16);

  const int HH = HIDC * HIDC;          // 262144
  const int HF = HIDC * 4 * HIDC;      // 1048576
  (void)HH; (void)HF;

  for (int i = 0; i < LYRC; ++i) {
    // weights -> transposed f16 (N x K), consumed as GEMM B tiles by the TDM
    convT(Wq + (size_t)i * HH, wq16, HIDC, HIDC);
    convT(Wk + (size_t)i * HH, wk16, HIDC, HIDC);
    convT(Wv + (size_t)i * HH, wv16, HIDC, HIDC);
    convT(Wo + (size_t)i * HH, wo16, HIDC, HIDC);
    convT(W1 + (size_t)i * HF, w116, HIDC, 4 * HIDC);
    convT(W2 + (size_t)i * HF, w216, 4 * HIDC, HIDC);

    // Q (scaled by dh^-0.5), K -> (B,H,S,dh); V -> (B,H,dh,S)
    gemm(h16, wq16, nullptr, nullptr, q16,  TOKC, HIDC, HIDC, 2, 0.125f);
    gemm(h16, wk16, nullptr, nullptr, k16,  TOKC, HIDC, HIDC, 2, 1.0f);
    gemm(h16, wv16, nullptr, nullptr, vT16, TOKC, HIDC, HIDC, 3, 1.0f);

    k_attn<<<dim3(NBC, HEADSC, BATCHC), dim3(128), 0, stream>>>(q16, k16, vT16, ao16);

    gemm(ao16, wo16, nullptr, yf32, nullptr, TOKC, HIDC, HIDC, 0, 1.0f);
    k_add_ln<<<dim3(TOKC), dim3(256), 0, stream>>>(yf32, hf32, h16,
                                                   ln1g + (size_t)i * HIDC,
                                                   ln1b + (size_t)i * HIDC);

    gemm(h16, w116, b1 + (size_t)i * 4 * HIDC, nullptr, mid16,
         TOKC, 4 * HIDC, HIDC, 1, 1.0f);
    gemm(mid16, w216, b2 + (size_t)i * HIDC, yf32, nullptr,
         TOKC, HIDC, 4 * HIDC, 0, 1.0f);
    k_add_ln<<<dim3(TOKC), dim3(256), 0, stream>>>(yf32, hf32, h16,
                                                   ln2g + (size_t)i * HIDC,
                                                   ln2b + (size_t)i * HIDC);
  }

  // ---- final projection (split-K over 768 chunks) + GELU + LN -----------
  k_proj<<<dim3(4, 768), dim3(128), 0, stream>>>(h16, projW, stage);
  k_reduce_stage<<<dim3(16), dim3(256), 0, stream>>>(stage, pre);
  k_out_ln<<<dim3(16), dim3(256), 0, stream>>>(pre, olng, olnb, out);
}
